// adjAttSubLayer_36644660970136
// MI455X (gfx1250) — compile-verified, matrix operand order checked by reading, not run on hardware
//
#include <hip/hip_runtime.h>

#define DEV __device__ __forceinline__

typedef unsigned short u16;
typedef unsigned int   u32;
typedef __attribute__((ext_vector_type(16))) __bf16 v16bf;
typedef __attribute__((ext_vector_type(8)))  float  v8f;
typedef __attribute__((ext_vector_type(8)))  u32    v8u;

constexpr int Bn = 4, S = 1024, Dm = 1024, H = 16, DK = 64;
constexpr float SCALE  = 0.125f;   // 1/sqrt(64)
constexpr float NEGV   = -1e9f;
constexpr float LNEPS  = 1e-5f;

// ---------- small helpers ----------
DEV u16 f2bf(float f) {              // RNE f32 -> bf16
  u32 u = __float_as_uint(f);
  u += 0x7FFFu + ((u >> 16) & 1u);
  return (u16)(u >> 16);
}
DEV u32 pack2(float lo, float hi) {
  return (u32)f2bf(lo) | ((u32)f2bf(hi) << 16);
}

// A/B fragment for v_wmma_f32_16x16x32_bf16, contraction along contiguous dim.
// p points at this lane's row, element k = off (off = 0 for lanes 0-15, 8 for 16-31).
// frag elems 0..7 = k off..off+7, elems 8..15 = k off+16..off+23 -> two 16B loads.
DEV v16bf frag_from_bf16(const u16* p) {
  uint4 a = *reinterpret_cast<const uint4*>(p);
  uint4 b = *reinterpret_cast<const uint4*>(p + 16);
  v8u u;
  u[0]=a.x; u[1]=a.y; u[2]=a.z; u[3]=a.w;
  u[4]=b.x; u[5]=b.y; u[6]=b.z; u[7]=b.w;
  return __builtin_bit_cast(v16bf, u);
}
DEV v8f wmma_bf16(v16bf a, v16bf b, v8f c) {
  return __builtin_amdgcn_wmma_f32_16x16x32_bf16(false, a, false, b, (short)0, c, false, false);
}
// reductions across a 16-lane half (xor masks 1,2,4,8 stay within each half of wave32)
DEV float rmax16(float v) {
#pragma unroll
  for (int o = 8; o >= 1; o >>= 1) v = fmaxf(v, __shfl_xor(v, o, 32));
  return v;
}
DEV float rsum16(float v) {
#pragma unroll
  for (int o = 8; o >= 1; o >>= 1) v += __shfl_xor(v, o, 32);
  return v;
}

// ---------- elementwise f32 -> bf16 (coalesced, 4 elems/thread) ----------
__global__ __launch_bounds__(256) void cvt_kernel(const float* __restrict__ in,
                                                  u16* __restrict__ out) {
  size_t i = ((size_t)blockIdx.x * 256 + threadIdx.x) * 4;
  float4 v = *reinterpret_cast<const float4*>(in + i);
  uint2 o;
  o.x = pack2(v.x, v.y);
  o.y = pack2(v.z, v.w);
  *reinterpret_cast<uint2*>(out + i) = o;
}

// ---------- tiled weight transpose: Wt[n*1024+k] = bf16(W[k*1024+n]) ----------
__global__ __launch_bounds__(256) void wt_kernel(const float* __restrict__ W,
                                                 u16* __restrict__ Wt) {
  __shared__ u16 tile[32][33];
  const int tx = threadIdx.x, ty = threadIdx.y;     // block (32,8)
  const int n0 = blockIdx.x * 32, k0 = blockIdx.y * 32;
#pragma unroll
  for (int j = 0; j < 32; j += 8)
    tile[ty + j][tx] = f2bf(W[(size_t)(k0 + ty + j) * 1024 + n0 + tx]);
  __syncthreads();
#pragma unroll
  for (int j = 0; j < 32; j += 8)
    Wt[(size_t)(n0 + ty + j) * 1024 + k0 + tx] = tile[tx][ty + j];
}

// ---------- tiled V transpose: VT[((b*H+h)*64+dk)*1024 + s] ----------
__global__ __launch_bounds__(256) void vt_kernel(const u16* __restrict__ Vb,
                                                 u16* __restrict__ VT) {
  __shared__ u16 tile[32][33];
  const int tx = threadIdx.x, ty = threadIdx.y;     // block (32,8)
  const int s0  = blockIdx.x * 32;                  // 32 tiles over S
  const int dk0 = (blockIdx.y & 1) * 32;            // 2 tiles over DK
  const int bh  = blockIdx.y >> 1;                  // 64 (b,h) pairs
  const int b = bh >> 4, h = bh & 15;
  const u16* src = Vb + ((size_t)b * S) * 1024 + h * 64;
#pragma unroll
  for (int j = 0; j < 32; j += 8)
    tile[ty + j][tx] = src[(size_t)(s0 + ty + j) * 1024 + dk0 + tx];
  __syncthreads();
  u16* dst = VT + ((size_t)bh * 64) * 1024;
#pragma unroll
  for (int j = 0; j < 32; j += 8)
    dst[(size_t)(dk0 + ty + j) * 1024 + s0 + tx] = tile[tx][ty + j];
}

// ---------- 4096x1024x1024 GEMM (bf16 in, WMMA), one wave per 16x64 tile ----------
__global__ __launch_bounds__(32) void gemm_kernel(const u16* __restrict__ A,   // [4096,1024] bf16
                                                  const u16* __restrict__ Wt,  // [N,K] bf16
                                                  void* __restrict__ Out, int out_f32,
                                                  const float* __restrict__ bias,
                                                  const float* __restrict__ resid) {
  const int lane = threadIdx.x;
  const int lm = lane & 15, half = lane >> 4, off = half * 8;
  const int m0 = blockIdx.x * 16;
  const int n0 = blockIdx.y * 64;

  v8f acc[4];
#pragma unroll
  for (int t = 0; t < 4; ++t)
#pragma unroll
    for (int e = 0; e < 8; ++e) acc[t][e] = 0.f;

  const u16* arow = A + (size_t)(m0 + lm) * 1024 + off;
  for (int k0 = 0; k0 < 1024; k0 += 32) {
    v16bf aF = frag_from_bf16(arow + k0);
#pragma unroll
    for (int t = 0; t < 4; ++t) {
      const u16* wrow = Wt + (size_t)(n0 + t * 16 + lm) * 1024 + k0 + off;
      acc[t] = wmma_bf16(aF, frag_from_bf16(wrow), acc[t]);
    }
  }
#pragma unroll
  for (int t = 0; t < 4; ++t)
#pragma unroll
    for (int r = 0; r < 8; ++r) {
      const int gm = m0 + r + half * 8;           // C/D: vgpr r -> row r (+8 for hi half)
      const int gn = n0 + t * 16 + lm;            // C/D: N = lane
      const size_t oi = (size_t)gm * 1024 + gn;
      float v = acc[t][r];
      if (out_f32) ((float*)Out)[oi] = v + bias[gn] + resid[oi];
      else         ((u16*)Out)[oi]   = f2bf(v);
    }
}

// ---------- attention: one wave per (b,h,16-row q tile) ----------
__global__ __launch_bounds__(32) void attn_kernel(const u16* __restrict__ Qb,
                                                  const u16* __restrict__ Kb,
                                                  const u16* __restrict__ VT,
                                                  const float* __restrict__ rel,
                                                  const int*  __restrict__ msk,
                                                  float* __restrict__ Pout,
                                                  u16*  __restrict__ attb) {
  __shared__ __align__(16) u16 p_lds[16][32];     // p tile: [q-row][k within 32-block]
  const int lane = threadIdx.x;
  const int lm = lane & 15, half = lane >> 4, off = half * 8;
  const int bid = blockIdx.x;
  const int qt = bid & 63;
  const int h  = (bid >> 6) & 15;
  const int b  = bid >> 10;
  const int q0 = qt * 16;

  const u16* Qbh  = Qb + ((size_t)b * S) * 1024 + (size_t)h * 64;
  const u16* Kbh  = Kb + ((size_t)b * S) * 1024 + (size_t)h * 64;
  const u16* VTbh = VT + ((size_t)(b * H + h) * DK) * 1024;
  const float* relb = rel + (size_t)b * S * S;
  const int*   mskb = msk + (size_t)b * S * S;
  float* Pbh = Pout + ((size_t)(b * H + h) * S) * S;

  const v16bf qF0 = frag_from_bf16(Qbh + (size_t)(q0 + lm) * 1024 + off);
  const v16bf qF1 = frag_from_bf16(Qbh + (size_t)(q0 + lm) * 1024 + 32 + off);

  float m[8], l[8];
#pragma unroll
  for (int r = 0; r < 8; ++r) { m[r] = -1e30f; l[r] = 0.f; }

  // ---- pass 1: online row max / sum-of-exp ----
  for (int kt = 0; kt < 64; ++kt) {
    {   // prefetch next tile's streams (global_prefetch_b8)
      const int kn = (kt + 1) & 63;
      const size_t pri = (size_t)(q0 + half * 8) * S + kn * 16 + lm;
      __builtin_prefetch(relb + pri, 0, 1);
      __builtin_prefetch(mskb + pri, 0, 1);
      __builtin_prefetch(Kbh + (size_t)(kn * 16 + lm) * 1024 + off, 0, 1);
    }
    const u16* krow = Kbh + (size_t)(kt * 16 + lm) * 1024;
    v8f sc;
#pragma unroll
    for (int e = 0; e < 8; ++e) sc[e] = 0.f;
    sc = wmma_bf16(qF0, frag_from_bf16(krow + off), sc);
    sc = wmma_bf16(qF1, frag_from_bf16(krow + 32 + off), sc);
    const int kc = kt * 16 + lm;
#pragma unroll
    for (int r = 0; r < 8; ++r) {
      const int gq = q0 + r + half * 8;
      const size_t ri = (size_t)gq * S + kc;
      float s = sc[r] * SCALE + relb[ri];
      s = (mskb[ri] != 0) ? NEGV : s;
      const float tm = rmax16(s);
      const float mn = fmaxf(m[r], tm);
      const float es = rsum16(__expf(s - mn));
      l[r] = l[r] * __expf(m[r] - mn) + es;
      m[r] = mn;
    }
  }

  v8f oacc[4];
#pragma unroll
  for (int t = 0; t < 4; ++t)
#pragma unroll
    for (int e = 0; e < 8; ++e) oacc[t][e] = 0.f;

  // ---- pass 2: recompute scores, emit normalized p, accumulate P@V ----
  for (int kt = 0; kt < 64; ++kt) {
    {
      const int kn = (kt + 1) & 63;
      const size_t pri = (size_t)(q0 + half * 8) * S + kn * 16 + lm;
      __builtin_prefetch(relb + pri, 0, 1);
      __builtin_prefetch(mskb + pri, 0, 1);
    }
    const u16* krow = Kbh + (size_t)(kt * 16 + lm) * 1024;
    v8f sc;
#pragma unroll
    for (int e = 0; e < 8; ++e) sc[e] = 0.f;
    sc = wmma_bf16(qF0, frag_from_bf16(krow + off), sc);
    sc = wmma_bf16(qF1, frag_from_bf16(krow + 32 + off), sc);
    const int kc = kt * 16 + lm;
#pragma unroll
    for (int r = 0; r < 8; ++r) {
      const int gq = q0 + r + half * 8;
      const size_t ri = (size_t)gq * S + kc;
      float s = sc[r] * SCALE + relb[ri];
      s = (mskb[ri] != 0) ? NEGV : s;
      const float pv = __expf(s - m[r]) / l[r];
      Pbh[ri] = pv;                                    // final softmax output
      p_lds[r + half * 8][(kt & 1) * 16 + lm] = f2bf(pv);
    }
    if (kt & 1) {                                      // have a full 16x32 p block
      asm volatile("s_wait_dscnt 0x0" ::: "memory");   // LDS writes visible (same wave)
      const v16bf pF = frag_from_bf16(&p_lds[lm][0] + off);   // A-layout read
      const int kj0 = (kt - 1) * 16;
#pragma unroll
      for (int t = 0; t < 4; ++t) {
        const u16* vrow = VTbh + (size_t)(t * 16 + lm) * 1024 + kj0 + off;
        oacc[t] = wmma_bf16(pF, frag_from_bf16(vrow), oacc[t]);
      }
    }
  }
  // att out, [b,s,h*dk] layout (feeds fc GEMM directly)
#pragma unroll
  for (int t = 0; t < 4; ++t)
#pragma unroll
    for (int r = 0; r < 8; ++r) {
      const int gq = q0 + r + half * 8;
      attb[((size_t)b * S + gq) * 1024 + h * 64 + t * 16 + lm] = f2bf(oacc[t][r]);
    }
}

// ---------- LayerNorm over rows of 1024 ----------
__global__ __launch_bounds__(256) void ln_kernel(const float* __restrict__ pre,
                                                 const float* __restrict__ g,
                                                 const float* __restrict__ be,
                                                 float* __restrict__ y) {
  __shared__ float sbuf[256];
  const int row = blockIdx.x, tid = threadIdx.x;
  const float* x = pre + (size_t)row * 1024;
  float a0 = x[tid], a1 = x[tid + 256], a2 = x[tid + 512], a3 = x[tid + 768];
  sbuf[tid] = a0 + a1 + a2 + a3;
  __syncthreads();
  for (int st = 128; st > 0; st >>= 1) { if (tid < st) sbuf[tid] += sbuf[tid + st]; __syncthreads(); }
  const float mu = sbuf[0] * (1.f / 1024.f);
  __syncthreads();
  float d0 = a0 - mu, d1 = a1 - mu, d2 = a2 - mu, d3 = a3 - mu;
  sbuf[tid] = d0 * d0 + d1 * d1 + d2 * d2 + d3 * d3;
  __syncthreads();
  for (int st = 128; st > 0; st >>= 1) { if (tid < st) sbuf[tid] += sbuf[tid + st]; __syncthreads(); }
  const float inv = rsqrtf(sbuf[0] * (1.f / 1024.f) + LNEPS);
  float* yr = y + (size_t)row * 1024;
  yr[tid]       = d0 * inv * g[tid]       + be[tid];
  yr[tid + 256] = d1 * inv * g[tid + 256] + be[tid + 256];
  yr[tid + 512] = d2 * inv * g[tid + 512] + be[tid + 512];
  yr[tid + 768] = d3 * inv * g[tid + 768] + be[tid + 768];
}

// ---------- host launcher ----------
extern "C" void kernel_launch(void* const* d_in, const int* in_sizes, int n_in,
                              void* d_out, int out_size, void* d_ws, size_t ws_size,
                              hipStream_t stream) {
  const float* query = (const float*)d_in[0];
  const float* key   = (const float*)d_in[1];
  const float* value = (const float*)d_in[2];
  const int*   maski = (const int*)  d_in[3];
  const float* rel   = (const float*)d_in[4];
  const float* Wq    = (const float*)d_in[5];
  const float* Wk    = (const float*)d_in[6];
  const float* Wv    = (const float*)d_in[7];
  const float* fcw   = (const float*)d_in[8];
  const float* fcb   = (const float*)d_in[9];
  const float* lng   = (const float*)d_in[10];
  const float* lnb   = (const float*)d_in[11];

  float* y = (float*)d_out;                         // [B,S,D]
  float* p = y + (size_t)Bn * S * Dm;               // [B,H,S,S]

  char* ws = (char*)d_ws;
  const size_t MB = 1024 * 1024;
  u16* wq_t = (u16*)(ws + 0 * MB);                  // 2MB each
  u16* wk_t = (u16*)(ws + 2 * MB);
  u16* wv_t = (u16*)(ws + 4 * MB);
  u16* wf_t = (u16*)(ws + 6 * MB);
  u16* Qb   = (u16*)(ws + 8 * MB);                  // 8MB each
  u16* Kb   = (u16*)(ws + 16 * MB);
  u16* Vb   = (u16*)(ws + 24 * MB);
  u16* VT   = (u16*)(ws + 32 * MB);
  u16* attb = (u16*)(ws + 40 * MB);
  float* pre = (float*)(ws + 48 * MB);              // 16MB
  u16* qbf  = (u16*)(ws + 64 * MB);                 // 8MB each
  u16* kbf  = (u16*)(ws + 72 * MB);
  u16* vbf  = (u16*)(ws + 80 * MB);

  // 0) activations f32 -> bf16 (once; keeps GEMM inner loop pure loads+wmma)
  cvt_kernel<<<4096, 256, 0, stream>>>(query, qbf);
  cvt_kernel<<<4096, 256, 0, stream>>>(key,   kbf);
  cvt_kernel<<<4096, 256, 0, stream>>>(value, vbf);

  // 1) bf16 transposed weights (coalesced tiled transpose)
  dim3 tg(32, 32), tb(32, 8);
  wt_kernel<<<tg, tb, 0, stream>>>(Wq,  wq_t);
  wt_kernel<<<tg, tb, 0, stream>>>(Wk,  wk_t);
  wt_kernel<<<tg, tb, 0, stream>>>(Wv,  wv_t);
  wt_kernel<<<tg, tb, 0, stream>>>(fcw, wf_t);

  // 2) projections (WMMA)
  dim3 gg(256, 16), bg(32);
  gemm_kernel<<<gg, bg, 0, stream>>>(qbf, wq_t, Qb, 0, nullptr, nullptr);
  gemm_kernel<<<gg, bg, 0, stream>>>(kbf, wk_t, Kb, 0, nullptr, nullptr);
  gemm_kernel<<<gg, bg, 0, stream>>>(vbf, wv_t, Vb, 0, nullptr, nullptr);

  // 3) V transpose for PV B-fragments (coalesced tiled transpose)
  vt_kernel<<<dim3(32, 128), tb, 0, stream>>>(Vb, VT);

  // 4) attention (WMMA QK^T + softmax + WMMA PV), writes p
  attn_kernel<<<Bn * H * (S / 16), 32, 0, stream>>>(Qb, Kb, VT, rel, maski, p, attb);

  // 5) fc + bias + residual (WMMA), then LayerNorm -> y
  gemm_kernel<<<gg, bg, 0, stream>>>(attb, wf_t, pre, 1, fcb, query);
  ln_kernel<<<Bn * S, 256, 0, stream>>>(pre, lng, lnb, y);
}
